// LSTMSkeleton_6390911337252
// MI455X (gfx1250) — matrix operand order repaired and checked by reading, not compiled, not run on hardware
//
#include <hip/hip_runtime.h>
#include <hip/hip_bf16.h>

// ---------------- problem constants ----------------
#define BB 32          // batch
#define TT 512         // time
#define JJOINT 25
#define NIN 50         // 2*J
#define DENC 256
#define HH 512         // hidden
#define G4 2048        // 4*H
#define MROWS (BB*TT)  // 16384

typedef __attribute__((ext_vector_type(16))) _Float16 v16h;
typedef __attribute__((ext_vector_type(8)))  float    v8f;
typedef __attribute__((ext_vector_type(2)))  _Float16 v2h;

// ---------------- workspace layout (bytes) ----------------
// total ~229 MB
#define OFF_ENC   ((size_t)0)                    // enc f16 [MROWS,256]          8,388,608
#define OFF_WIH0  ((size_t)8388608)              // f16 [2,2048,256]             2,097,152
#define OFF_WHH0  ((size_t)10485760)             // f16 [2,2048,512]             4,194,304
#define OFF_WIH1  ((size_t)14680064)             // f16 [2,2048,1024]            8,388,608
#define OFF_WHH1  ((size_t)23068672)             // f16 [2,2048,512]             4,194,304
#define OFF_XG    ((size_t)27262976)             // f16 [2,MROWS,2048]         134,217,728
#define OFF_Y0    ((size_t)161480704)            // f16 [B,T,1024]              33,554,432
#define OFF_Y1    ((size_t)195035136)            // f16 [B,T,1024]              33,554,432
#define OFF_H     ((size_t)228589568)            // f16 [2 dir][2 phase][B,H]      131,072
#define OFF_BAR   ((size_t)228720640)            // int [T]                          2,048

// ---------------- CDNA5 async global->LDS helpers ----------------
// global_load_async_to_lds_b128: VDST = per-lane LDS byte address, VADDR = 64b global addr.
// Tracked by ASYNCcnt; waited with s_wait_asynccnt.
__device__ __forceinline__ unsigned lds_off(const void* p) {
  // generic -> LDS address-space cast, then take the 32-bit LDS byte offset
  return (unsigned)(unsigned long long)(const __attribute__((address_space(3))) char*)p;
}
__device__ __forceinline__ void async_copy_b128(unsigned lds_byte, const void* gsrc) {
  asm volatile("global_load_async_to_lds_b128 %0, %1, off"
               :: "v"(lds_byte), "v"(gsrc)
               : "memory");
}
__device__ __forceinline__ void async_wait0() {
  asm volatile("s_wait_asynccnt 0" ::: "memory");
}

// ---------------- f32 -> f16 convert ----------------
__global__ void k_cvt(const float* __restrict__ src, _Float16* __restrict__ dst, int n) {
  int i = blockIdx.x * blockDim.x + threadIdx.x;
  if (i < n) dst[i] = (_Float16)src[i];
}

// ---------------- encoder: conf-mask + relu(x @ W^T + b) -> f16 ----------------
__global__ void __launch_bounds__(256) k_encode(const float* __restrict__ feat,
                                                const float* __restrict__ W,
                                                const float* __restrict__ bias,
                                                _Float16* __restrict__ enc) {
  __shared__ float xin[NIN];
  int row = blockIdx.x;                       // b*T + t
  const float* f = feat + (size_t)row * (JJOINT * 3);
  int tid = threadIdx.x;
  if (tid < JJOINT) {
    float c = f[tid * 3 + 2] > 0.1f ? 1.0f : 0.0f;
    xin[tid * 2 + 0] = f[tid * 3 + 0] * c;
    xin[tid * 2 + 1] = f[tid * 3 + 1] * c;
  }
  __syncthreads();
  float acc = bias[tid];
  const float* w = W + tid * NIN;
#pragma unroll
  for (int k = 0; k < NIN; ++k) acc += w[k] * xin[k];
  acc = acc > 0.0f ? acc : 0.0f;
  enc[(size_t)row * DENC + tid] = (_Float16)acc;
}

// ---------------- input-projection GEMM (WMMA f16) ----------------
// xg[dir][m][n] = sum_k A[m,k]*W[dir][n,k] + bias[dir][n]   (stored f16)
// grid: (M/64, 2048/64, 2), block 128 (4 waves, each one 16x64 strip)
__global__ void __launch_bounds__(128) k_gemm_xg(const _Float16* __restrict__ A,
                                                 const _Float16* __restrict__ Wf,
                                                 const float* __restrict__ bias,
                                                 _Float16* __restrict__ xg,
                                                 int M, int K) {
  int lane = threadIdx.x & 31;
  int wave = threadIdx.x >> 5;
  int mt  = blockIdx.x * 4 + wave;      // 16-row tile index
  int n0  = blockIdx.y * 64;
  int dir = blockIdx.z;
  const _Float16* Wd = Wf + (size_t)dir * G4 * K;
  int m0 = mt * 16;

  v8f acc[4] = {};
  int am  = m0 + (lane & 15);
  int akb = (lane >> 4) << 4;           // A: half-lanes 0/16 k-offset
  int bn  = lane & 15;
  int bkb = (lane >> 4) << 4;           // B: half-lanes 0/16 k-offset

  for (int k0 = 0; k0 < K; k0 += 32) {
    v16h a;
    const _Float16* ap = A + (size_t)am * K + k0;
#pragma unroll
    for (int i = 0; i < 4; ++i) {
      v2h lo = *(const v2h*)(ap + akb + 2 * i);
      v2h hi = *(const v2h*)(ap + 8 + akb + 2 * i);
      a[2 * i] = lo.x; a[2 * i + 1] = lo.y;
      a[8 + 2 * i] = hi.x; a[8 + 2 * i + 1] = hi.y;
    }
#pragma unroll
    for (int nt = 0; nt < 4; ++nt) {
      v16h bf;
      const _Float16* bp = Wd + (size_t)(n0 + nt * 16 + bn) * K + k0 + bkb;
#pragma unroll
      for (int i = 0; i < 8; ++i) {
        v2h t = *(const v2h*)(bp + 2 * i);
        bf[2 * i] = t.x; bf[2 * i + 1] = t.y;
      }
      acc[nt] = __builtin_amdgcn_wmma_f32_16x16x32_f16(
          false, a, false, bf, (short)0, acc[nt], false, false);
    }
  }
  // epilogue: C layout — VGPR i holds row (lane<16 ? i : 8+i), col lane&15
  int rlo  = (lane < 16) ? 0 : 8;
  int ccol = lane & 15;
#pragma unroll
  for (int nt = 0; nt < 4; ++nt) {
    int n = n0 + nt * 16 + ccol;
    float bv = bias[dir * G4 + n];
#pragma unroll
    for (int i = 0; i < 8; ++i) {
      int m = m0 + rlo + i;
      xg[((size_t)dir * M + m) * G4 + n] = (_Float16)(acc[nt][i] + bv);
    }
  }
}

// ---------------- persistent BiLSTM recurrence ----------------
// 32 WGs: dir = wg>>4, slot = wg&15, each owns 32 hidden units.
// W_hh slice (128 rows x 512) resident in LDS; h double-buffered in global.
#define HP 520     // h LDS pitch (halves);  HP*2 = 1040 B = 65 x 16B
#define WPI 520    // W LDS pitch (halves)
#define GP 132

__global__ void __launch_bounds__(256) k_lstm_rec(const _Float16* __restrict__ whh,  // [2,2048,512]
                                                  const _Float16* __restrict__ xg,   // [2,MROWS,2048]
                                                  const int* __restrict__ lengths,
                                                  _Float16* __restrict__ Y,          // [B,T,1024]
                                                  _Float16* h_g,                     // [2][2][B][H]
                                                  int* bar) {
  __shared__ alignas(16) _Float16 wbuf[128 * WPI];
  __shared__ alignas(16) _Float16 hbuf[BB * HP];
  __shared__ float    gbuf[BB * GP];
  __shared__ float    cbuf[BB * 32];
  __shared__ int      lbuf[BB];

  int tid  = threadIdx.x;
  int lane = tid & 31;
  int wave = tid >> 5;
  int wg   = blockIdx.x;
  int dir  = wg >> 4;
  int slot = wg & 15;
  int j0   = slot * 32;
  int nwg  = gridDim.x;

  unsigned wbuf_base = lds_off(&wbuf[0]);
  unsigned hbuf_base = lds_off(&hbuf[0]);

  // ---- async preload of resident W_hh slice ----
  // LDS row r (r>>5 = gate, r&31 = jj) <- global row g*512 + j0 + jj ; rows are 1024B = 64 x 16B
  {
    const char* wsrc = (const char*)(whh + (size_t)dir * G4 * HH);
#pragma unroll
    for (int i = 0; i < 32; ++i) {          // 8192 chunks / 256 threads
      int idx = tid + i * 256;
      int r = idx >> 6;
      int c = idx & 63;
      int n = ((r >> 5) * HH) + j0 + (r & 31);
      async_copy_b128(wbuf_base + (unsigned)(r * (WPI * 2) + c * 16),
                      wsrc + (size_t)n * 1024 + c * 16);
    }
  }
  if (tid < BB) lbuf[tid] = lengths[tid];
  for (int i = tid; i < BB * 32; i += 256) cbuf[i] = 0.0f;
  async_wait0();
  __syncthreads();

  int mt = wave & 1;      // 8 waves = 2 M-tiles x 4 wave-columns; each wave does 2 of 8 N-tiles
  int nt = wave >> 1;

  for (int s = 0; s < TT; ++s) {
    // ---- async broadcast of h(s-1) into LDS (read phase s&1) ----
    {
      const char* hsrc =
          (const char*)(h_g + ((size_t)dir * 2 + (size_t)(s & 1)) * BB * HH);
#pragma unroll
      for (int i = 0; i < 8; ++i) {          // 2048 chunks / 256 threads
        int idx = tid + i * 256;
        int r = idx >> 6;                    // batch row (64 x 16B per row)
        int c = idx & 63;
        async_copy_b128(hbuf_base + (unsigned)(r * (HP * 2) + c * 16),
                        hsrc + (size_t)idx * 16);
      }
    }
    async_wait0();
    __syncthreads();

    // ---- gates = h @ Wslice^T : each wave does 2 (16x16) output tiles ----
    int am  = mt * 16 + (lane & 15);
    int akb = (lane >> 4) << 4;
    int bkb = (lane >> 4) << 4;
#pragma unroll
    for (int half = 0; half < 2; ++half) {
      int ntile = nt * 2 + half;                 // 0..7 -> gate cols ntile*16..+15
      v8f acc = {};
      int brow = ntile * 16 + (lane & 15);
      for (int k0 = 0; k0 < HH; k0 += 32) {
        v16h a, bf;
        const _Float16* ap = &hbuf[am * HP + k0];
#pragma unroll
        for (int i = 0; i < 4; ++i) {
          v2h lo = *(const v2h*)(ap + akb + 2 * i);
          v2h hi = *(const v2h*)(ap + 8 + akb + 2 * i);
          a[2 * i] = lo.x; a[2 * i + 1] = lo.y;
          a[8 + 2 * i] = hi.x; a[8 + 2 * i + 1] = hi.y;
        }
        const _Float16* bp = &wbuf[brow * WPI + k0 + bkb];
#pragma unroll
        for (int i = 0; i < 8; ++i) {
          v2h t = *(const v2h*)(bp + 2 * i);
          bf[2 * i] = t.x; bf[2 * i + 1] = t.y;
        }
        acc = __builtin_amdgcn_wmma_f32_16x16x32_f16(
            false, a, false, bf, (short)0, acc, false, false);
      }
      int rbase = mt * 16 + ((lane < 16) ? 0 : 8);
      int col   = ntile * 16 + (lane & 15);
#pragma unroll
      for (int i = 0; i < 8; ++i) gbuf[(rbase + i) * GP + col] = acc[i];
    }
    __syncthreads();

    // ---- LSTM cell for 32 batch x 32 hidden (4 cells/thread) ----
    _Float16* h_wr = h_g + ((size_t)dir * 2 + (size_t)((s + 1) & 1)) * BB * HH;
#pragma unroll
    for (int q = 0; q < 4; ++q) {
      int cell = (tid << 2) + q;
      int m  = cell >> 5;
      int jj = cell & 31;
      int len = lbuf[m];
      bool valid = s < len;
      int tq = (dir == 0) ? s : (valid ? (len - 1 - s) : s);
      size_t xoff = ((size_t)dir * MROWS + (size_t)m * TT + tq) * G4 + j0 + jj;
      float gi = gbuf[m * GP + jj]        + (float)xg[xoff];
      float gf = gbuf[m * GP + 32 + jj]   + (float)xg[xoff + 512];
      float gg = gbuf[m * GP + 64 + jj]   + (float)xg[xoff + 1024];
      float go = gbuf[m * GP + 96 + jj]   + (float)xg[xoff + 1536];
      float ig = 1.0f / (1.0f + __expf(-gi));
      float fg = 1.0f / (1.0f + __expf(-gf));
      float og = 1.0f / (1.0f + __expf(-go));
      float e2 = __expf(2.0f * gg);
      float gt = (e2 - 1.0f) / (e2 + 1.0f);
      float cn = fg * cbuf[m * 32 + jj] + ig * gt;
      float ec = __expf(2.0f * cn);
      float hn = og * (ec - 1.0f) / (ec + 1.0f);
      _Float16 hprev = hbuf[m * HP + j0 + jj];
      _Float16 hout  = valid ? (_Float16)hn : hprev;      // keep h when masked
      if (valid) cbuf[m * 32 + jj] = cn;
      h_wr[(size_t)m * HH + j0 + jj] = hout;
      size_t yoff = ((size_t)m * TT + tq) * 1024 + (size_t)dir * 512 + j0 + jj;
      Y[yoff] = valid ? (_Float16)hn : (_Float16)0.0f;

      // prefetch next step's xg gate slices (one thread per (m) row-slice)
      if (s + 1 < TT && (cell & 31) == 0) {
        bool v2 = (s + 1) < len;
        int tq2 = (dir == 0) ? (s + 1) : (v2 ? (len - 2 - s) : (s + 1));
        const _Float16* px =
            xg + ((size_t)dir * MROWS + (size_t)m * TT + tq2) * G4 + j0;
        __builtin_prefetch(px, 0, 3);
        __builtin_prefetch(px + 512, 0, 3);
        __builtin_prefetch(px + 1024, 0, 3);
        __builtin_prefetch(px + 1536, 0, 3);
      }
    }

    // ---- grid-wide step barrier (arrive + spin); double-buffered h => 1 barrier/step ----
    __threadfence();
    __syncthreads();
    if (tid == 0) {
      atomicAdd(&bar[s], 1);
      volatile int* vb = bar;
      while (vb[s] < nwg) { __builtin_amdgcn_s_sleep(1); }
      __threadfence();
    }
    __syncthreads();
  }
}

// ---------------- sum pooling / lengths ----------------
__global__ void __launch_bounds__(256) k_pool(const _Float16* __restrict__ Y,
                                              const int* __restrict__ lengths,
                                              float* __restrict__ out) {
  int c = blockIdx.y * 256 + threadIdx.x;
  int b = blockIdx.x;
  float acc = 0.0f;
  for (int t = 0; t < TT; ++t) acc += (float)Y[((size_t)b * TT + t) * 1024 + c];
  out[(size_t)b * 1024 + c] = acc / (float)lengths[b];
}

// ---------------- launcher ----------------
extern "C" void kernel_launch(void* const* d_in, const int* in_sizes, int n_in,
                              void* d_out, int out_size, void* d_ws, size_t ws_size,
                              hipStream_t stream) {
  const float* features = (const float*)d_in[0];
  const int*   lengths  = (const int*)d_in[1];
  const float* enc_W    = (const float*)d_in[2];
  const float* enc_b    = (const float*)d_in[3];
  const float* w_ih_l0  = (const float*)d_in[4];
  const float* w_hh_l0  = (const float*)d_in[5];
  const float* b_l0     = (const float*)d_in[6];
  const float* w_ih_l1  = (const float*)d_in[7];
  const float* w_hh_l1  = (const float*)d_in[8];
  const float* b_l1     = (const float*)d_in[9];

  char* ws = (char*)d_ws;
  _Float16* enc16 = (_Float16*)(ws + OFF_ENC);
  _Float16* wih0  = (_Float16*)(ws + OFF_WIH0);
  _Float16* whh0  = (_Float16*)(ws + OFF_WHH0);
  _Float16* wih1  = (_Float16*)(ws + OFF_WIH1);
  _Float16* whh1  = (_Float16*)(ws + OFF_WHH1);
  _Float16* xg    = (_Float16*)(ws + OFF_XG);
  _Float16* y0    = (_Float16*)(ws + OFF_Y0);
  _Float16* y1    = (_Float16*)(ws + OFF_Y1);
  _Float16* hbufg = (_Float16*)(ws + OFF_H);
  int*      barp  = (int*)(ws + OFF_BAR);

  // weight converts (parallel, off the critical path)
  k_cvt<<<(2 * G4 * DENC + 255) / 256, 256, 0, stream>>>(w_ih_l0, wih0, 2 * G4 * DENC);
  k_cvt<<<(2 * G4 * HH + 255) / 256, 256, 0, stream>>>(w_hh_l0, whh0, 2 * G4 * HH);
  k_cvt<<<(2 * G4 * 1024 + 255) / 256, 256, 0, stream>>>(w_ih_l1, wih1, 2 * G4 * 1024);
  k_cvt<<<(2 * G4 * HH + 255) / 256, 256, 0, stream>>>(w_hh_l1, whh1, 2 * G4 * HH);

  // encoder
  k_encode<<<MROWS, 256, 0, stream>>>(features, enc_W, enc_b, enc16);

  // ---- layer 0 ----
  hipMemsetAsync(hbufg, 0, (size_t)2 * 2 * BB * HH * sizeof(_Float16), stream);
  hipMemsetAsync(barp, 0, (size_t)TT * sizeof(int), stream);
  k_gemm_xg<<<dim3(MROWS / 64, G4 / 64, 2), 128, 0, stream>>>(enc16, wih0, b_l0, xg, MROWS, DENC);
  k_lstm_rec<<<32, 256, 0, stream>>>(whh0, xg, lengths, y0, hbufg, barp);

  // ---- layer 1 ----
  hipMemsetAsync(hbufg, 0, (size_t)2 * 2 * BB * HH * sizeof(_Float16), stream);
  hipMemsetAsync(barp, 0, (size_t)TT * sizeof(int), stream);
  k_gemm_xg<<<dim3(MROWS / 64, G4 / 64, 2), 128, 0, stream>>>(y0, wih1, b_l1, xg, MROWS, 1024);
  k_lstm_rec<<<32, 256, 0, stream>>>(whh1, xg, lengths, y1, hbufg, barp);

  // pooling
  k_pool<<<dim3(BB, 4), 256, 0, stream>>>(y1, lengths, (float*)d_out);
}
